// Quantize_64561948393617
// MI455X (gfx1250) — matrix operand order repaired and checked by reading, not compiled, not run on hardware
//
#include <hip/hip_runtime.h>
#include <hip/hip_bf16.h>
#include <stdint.h>

// ---------------- problem constants ----------------
constexpr int B_  = 8;
constexpr int T_  = 4096;
constexpr int C_  = 768;
constexpr int G_  = 2;
constexpr int V_  = 320;
constexpr int GV_ = G_ * V_;      // 640
constexpr int CG_ = C_ / G_;      // 384
constexpr float EPS_ = 1e-5f;

typedef __attribute__((ext_vector_type(16))) _Float16 v16h;
typedef __attribute__((ext_vector_type(8)))  float    v8f;

union Frag16 {                 // 8 VGPRs = 16 halves
    v16h     v;
    uint32_t u[8];
    uint4    q[2];
    _Float16 h[16];
};

union U4H { uint4 u; _Float16 h[8]; };

__device__ __forceinline__ uint32_t pack_h2(float x, float y) {
    union { _Float16 h[2]; uint32_t u; } t;
    t.h[0] = (_Float16)x; t.h[1] = (_Float16)y;
    return t.u;
}

// gumbel-from-logits transform: y = (x + (-log(-log(x+eps)+eps))) / TEMP
__device__ __forceinline__ float gumbel_y(float x) {
    float lg = __logf(x + EPS_);
    float gm = -__logf(EPS_ - lg);
    return (x + gm) * 0.5f;                 // TEMP = 2
}

// ---------------- kernel 0: fp32 -> f16 weight/codebook conversion ----------
constexpr int NW_  = GV_ * C_;        // 491520
constexpr int NCB_ = G_ * CG_ * V_;   // 245760

__global__ __launch_bounds__(256)
void cvt_f16_kernel(const float* __restrict__ Wf, const float* __restrict__ cbf,
                    _Float16* __restrict__ Wh, _Float16* __restrict__ cbh) {
    int i = blockIdx.x * 256 + threadIdx.x;
    if (i < NW_)  Wh[i]  = (_Float16)Wf[i];
    if (i < NCB_) cbh[i] = (_Float16)cbf[i];
}

// ---------------- kernel 1: logits GEMM (X @ W^T + b), write [B,GV,T] -------
// macro-tile 128t x 128n, 8 waves, each wave 32t x 64n = 8 WMMA frags.
constexpr int KB  = 32;    // WMMA K
constexpr int LDA = 40;    // LDS half-stride (odd word stride -> conflict free)
constexpr int LDC = 66;    // LDS float-stride for 64-wide C staging chunks

__global__ __launch_bounds__(256)
void gemm_logits_kernel(const float* __restrict__ inp,    // [B,T,C]
                        const _Float16* __restrict__ Wh,  // [GV,C] f16
                        const float* __restrict__ bias,   // [GV]
                        float* __restrict__ logits)       // [B,GV,T]
{
    __shared__ _Float16 Ah[128 * LDA];     // 128t x 32k  (10,240 B)
    __shared__ _Float16 Bh[128 * LDA];     // 128n x 32k  (10,240 B)
    __shared__ float    Cs[128 * LDC];     // 128t x 64n  (33,792 B)

    const int tid  = threadIdx.x;
    const int wv   = tid >> 5;
    const int lane = tid & 31;
    const int m    = lane & 15;
    const int hi   = lane >> 4;

    const int tq4 = wv & 3;    // wave t-base = tq4*32 (2 tiles of 16)
    const int nq  = wv >> 2;   // wave n-base = nq*64  (4 tiles of 16)

    const int t0 = blockIdx.x * 128;
    const int n0 = blockIdx.y * 128;
    const int b  = blockIdx.z;

    const float*    Ain = inp + ((size_t)b * T_ + t0) * C_;
    const _Float16* Bin = Wh + (size_t)n0 * C_;

    v8f acc[8];
#pragma unroll
    for (int i = 0; i < 8; ++i)
#pragma unroll
        for (int r = 0; r < 8; ++r) acc[i][r] = 0.0f;

    for (int k0 = 0; k0 < C_; k0 += KB) {
        // ---- issue ALL tile loads first (graduated loadcnt waits) ----
        float4 fa[4];
#pragma unroll
        for (int it = 0; it < 4; ++it) {
            int s = it * 256 + tid;
            int r = s >> 3, j = s & 7;
            fa[it] = *(const float4*)(Ain + (size_t)r * C_ + k0 + j * 4);
        }
        uint4 wb[2];
#pragma unroll
        for (int it = 0; it < 2; ++it) {
            int s = it * 256 + tid;
            int n = s >> 2, qq = s & 3;
            wb[it] = *(const uint4*)(Bin + (size_t)n * C_ + k0 + qq * 8);
        }
        if (k0 + KB < C_)   // CDNA5 global_prefetch of next k-slice of inputs
            __builtin_prefetch((const void*)(Ain + (size_t)(tid >> 1) * C_ + k0 + KB + (tid & 1) * 16), 0, 1);

        // ---- convert / store to LDS ----
#pragma unroll
        for (int it = 0; it < 4; ++it) {
            int s = it * 256 + tid;
            int r = s >> 3, j = s & 7;
            uint2 p; p.x = pack_h2(fa[it].x, fa[it].y); p.y = pack_h2(fa[it].z, fa[it].w);
            *(uint2*)&Ah[r * LDA + j * 4] = p;
        }
#pragma unroll
        for (int it = 0; it < 2; ++it) {
            int s = it * 256 + tid;
            int n = s >> 2, qq = s & 3;
            *(uint4*)&Bh[n * LDA + qq * 8] = wb[it];
        }
        __syncthreads();

        // fragments: 2 A (t-tiles) + 4 B (n-tiles), then 8 WMMAs
        Frag16 a[2], bf[4];
#pragma unroll
        for (int ti = 0; ti < 2; ++ti) {
            const int row = tq4 * 32 + ti * 16 + m;
            a[ti].q[0] = *(const uint4*)&Ah[row * LDA + (hi << 3)];
            a[ti].q[1] = *(const uint4*)&Ah[row * LDA + 16 + (hi << 3)];
        }
#pragma unroll
        for (int nt = 0; nt < 4; ++nt) {
            const int row = nq * 64 + nt * 16 + m;
            bf[nt].q[0] = *(const uint4*)&Bh[row * LDA + (hi << 3)];
            bf[nt].q[1] = *(const uint4*)&Bh[row * LDA + 16 + (hi << 3)];
        }
#pragma unroll
        for (int ti = 0; ti < 2; ++ti)
#pragma unroll
            for (int nt = 0; nt < 4; ++nt)
                acc[ti * 4 + nt] = __builtin_amdgcn_wmma_f32_16x16x32_f16(
                    false, a[ti].v, false, bf[nt].v, (short)0, acc[ti * 4 + nt], false, false);
        __syncthreads();
    }

    // epilogue: bias + transpose-store, 2 chunks of 64 n columns
    float* Lout = logits + (size_t)b * GV_ * T_;
#pragma unroll
    for (int h = 0; h < 2; ++h) {
        if (h) __syncthreads();
        if (nq == h) {
#pragma unroll
            for (int nt = 0; nt < 4; ++nt) {
                const int nl = nt * 16 + m;                  // chunk-local col
                const float bv = bias[n0 + h * 64 + nl];
#pragma unroll
                for (int ti = 0; ti < 2; ++ti)
#pragma unroll
                    for (int r = 0; r < 8; ++r) {
                        int tl = tq4 * 32 + ti * 16 + r + hi * 8;
                        Cs[tl * LDC + nl] = acc[ti * 4 + nt][r] + bv;
                    }
            }
        }
        __syncthreads();
        // coalesced store: logits[b][n0+h*64+n][t0 .. t0+127]
#pragma unroll
        for (int it = 0; it < 8; ++it) {
            int s = it * 256 + tid;
            int n = s >> 5, tqq = s & 31;
            float4 o;
            o.x = Cs[(tqq * 4 + 0) * LDC + n];
            o.y = Cs[(tqq * 4 + 1) * LDC + n];
            o.z = Cs[(tqq * 4 + 2) * LDC + n];
            o.w = Cs[(tqq * 4 + 3) * LDC + n];
            *(float4*)(Lout + (size_t)(n0 + h * 64 + n) * T_ + t0 + tqq * 4) = o;
        }
    }
}

// ---------------- kernel 2: gumbel softmax over V + argmax ------------------
__global__ __launch_bounds__(256)
void softmax_kernel(const float* __restrict__ logits,  // [B*G, V, T]
                    _Float16* __restrict__ prob,       // [B*G, V, T] f16
                    int* __restrict__ amax)            // [B*G, T]
{
    const int t  = blockIdx.x * 256 + threadIdx.x;
    const int bg = blockIdx.y;
    const float* Lp = logits + (size_t)bg * V_ * T_ + t;

    float ymax = -3.4e38f; int arg = 0;
#pragma unroll 4
    for (int v = 0; v < V_; ++v) {
        float y = gumbel_y(Lp[(size_t)v * T_]);
        if (y > ymax) { ymax = y; arg = v; }
    }
    float s = 0.0f;
#pragma unroll 4
    for (int v = 0; v < V_; ++v)
        s += __expf(gumbel_y(Lp[(size_t)v * T_]) - ymax);
    const float inv = 1.0f / s;
    _Float16* Pp = prob + (size_t)bg * V_ * T_ + t;
#pragma unroll 4
    for (int v = 0; v < V_; ++v)
        Pp[(size_t)v * T_] = (_Float16)(__expf(gumbel_y(Lp[(size_t)v * T_]) - ymax) * inv);
    amax[(size_t)bg * T_ + t] = arg;
}

// ---------------- kernel 3: cb x prob GEMM + straight-through combine -------
// macro-tile 128c x 128t per (b,g); wave = 32c x 64t = 8 WMMA frags.
__global__ __launch_bounds__(256)
void gemm_codes_kernel(const _Float16* __restrict__ cbh,  // [G,CG,V] f16
                       const _Float16* __restrict__ prob, // [B*G,V,T] f16
                       const float* __restrict__ cbf,     // [G,CG,V] f32
                       const int* __restrict__ amax,      // [B*G,T]
                       float* __restrict__ out)           // [B,T,C]
{
    __shared__ _Float16 Ah[128 * LDA];     // 128c x 32v
    __shared__ _Float16 Bt[128 * LDA];     // 128t x 32v (transposed prob tile)
    __shared__ float    Cs[128 * LDC];     // 128t x 64c chunk

    const int tid  = threadIdx.x;
    const int wv   = tid >> 5;
    const int lane = tid & 31;
    const int m    = lane & 15;
    const int hi   = lane >> 4;

    const int mq = wv & 3;     // wave c-base = mq*32 (2 tiles)
    const int nq = wv >> 2;    // wave t-base = nq*64 (4 tiles)

    const int t0 = blockIdx.x * 128;
    const int c0 = blockIdx.y * 128;
    const int bg = blockIdx.z;
    const int b  = bg >> 1;
    const int g  = bg & 1;

    const _Float16* Ain = cbh + ((size_t)g * CG_ + c0) * V_;
    const _Float16* Bin = prob + (size_t)bg * V_ * T_;

    v8f acc[8];
#pragma unroll
    for (int i = 0; i < 8; ++i)
#pragma unroll
        for (int r = 0; r < 8; ++r) acc[i][r] = 0.0f;

    for (int kv = 0; kv < V_; kv += KB) {
        // ---- issue ALL tile loads first ----
        uint4 ca[2];
#pragma unroll
        for (int it = 0; it < 2; ++it) {
            int s = it * 256 + tid;
            int c = s >> 2, qq = s & 3;
            ca[it] = *(const uint4*)(Ain + (size_t)c * V_ + kv + qq * 8);
        }
        U4H pb[2];
#pragma unroll
        for (int it = 0; it < 2; ++it) {
            int s  = it * 256 + tid;
            int kk = s & 31;        // v within slice
            int qq = s >> 5;        // t octet 0..15
            pb[it].u = *(const uint4*)(Bin + (size_t)(kv + kk) * T_ + t0 + qq * 8);
        }
        if (kv + KB < V_)   // prefetch next prob slice
            __builtin_prefetch((const void*)(Bin + (size_t)(kv + KB + (tid & 31)) * T_ + t0 + (tid >> 5) * 16), 0, 1);

        // ---- store to LDS (B transposed: lanes walk v -> consecutive halves) ----
#pragma unroll
        for (int it = 0; it < 2; ++it) {
            int s = it * 256 + tid;
            int c = s >> 2, qq = s & 3;
            *(uint4*)&Ah[c * LDA + qq * 8] = ca[it];
        }
#pragma unroll
        for (int it = 0; it < 2; ++it) {
            int s  = it * 256 + tid;
            int kk = s & 31;
            int qq = s >> 5;
#pragma unroll
            for (int j = 0; j < 8; ++j)
                Bt[(qq * 8 + j) * LDA + kk] = pb[it].h[j];
        }
        __syncthreads();

        Frag16 a[2], bf[4];
#pragma unroll
        for (int ti = 0; ti < 2; ++ti) {
            const int row = mq * 32 + ti * 16 + m;
            a[ti].q[0] = *(const uint4*)&Ah[row * LDA + (hi << 3)];
            a[ti].q[1] = *(const uint4*)&Ah[row * LDA + 16 + (hi << 3)];
        }
#pragma unroll
        for (int nt = 0; nt < 4; ++nt) {
            const int row = nq * 64 + nt * 16 + m;           // t row
            bf[nt].q[0] = *(const uint4*)&Bt[row * LDA + (hi << 3)];
            bf[nt].q[1] = *(const uint4*)&Bt[row * LDA + 16 + (hi << 3)];
        }
#pragma unroll
        for (int ti = 0; ti < 2; ++ti)
#pragma unroll
            for (int nt = 0; nt < 4; ++nt)
                acc[ti * 4 + nt] = __builtin_amdgcn_wmma_f32_16x16x32_f16(
                    false, a[ti].v, false, bf[nt].v, (short)0, acc[ti * 4 + nt], false, false);
        __syncthreads();
    }

    // epilogue: 2 chunks of 64 c columns, STE combine on writeout
    const float* cbrow = cbf + (size_t)g * CG_ * V_;
#pragma unroll
    for (int h = 0; h < 2; ++h) {
        if (h) __syncthreads();
        if ((mq >> 1) == h) {
#pragma unroll
            for (int nt = 0; nt < 4; ++nt) {
                const int tl = nq * 64 + nt * 16 + m;
#pragma unroll
                for (int ti = 0; ti < 2; ++ti)
#pragma unroll
                    for (int r = 0; r < 8; ++r) {
                        int cl = (mq & 1) * 32 + ti * 16 + r + hi * 8;  // chunk-local
                        Cs[tl * LDC + cl] = acc[ti * 4 + nt][r];
                    }
            }
        }
        __syncthreads();
        // out[b][t][g*CG + c0 + h*64 + c] = (hard - soft) + soft
#pragma unroll
        for (int it = 0; it < 8; ++it) {
            int s = it * 256 + tid;
            int tr = s >> 4, qq = s & 15;
            int tg = t0 + tr;
            int vstar = amax[(size_t)bg * T_ + tg];
            float4 o;
#pragma unroll
            for (int i = 0; i < 4; ++i) {
                int cl = qq * 4 + i;                        // 0..63 chunk-local
                float soft = Cs[tr * LDC + cl];
                float hard = cbrow[(size_t)(c0 + h * 64 + cl) * V_ + vstar];
                (&o.x)[i] = (hard - soft) + soft;
            }
            *(float4*)(out + ((size_t)b * T_ + tg) * C_ + g * CG_ + c0 + h * 64 + qq * 4) = o;
        }
    }
}

// ---------------- launcher --------------------------------------------------
extern "C" void kernel_launch(void* const* d_in, const int* in_sizes, int n_in,
                              void* d_out, int out_size, void* d_ws, size_t ws_size,
                              hipStream_t stream) {
    (void)in_sizes; (void)n_in; (void)out_size; (void)ws_size;
    const float* inp  = (const float*)d_in[0];   // [B,T,C]
    const float* Wf   = (const float*)d_in[1];   // [GV,C]
    const float* bias = (const float*)d_in[2];   // [GV]
    const float* cbf  = (const float*)d_in[3];   // [1,G,CG,V]

    float* out    = (float*)d_out;                          // [B,T,C]
    float* logits = out + (size_t)B_ * T_ * C_;             // [B,G,V,T]

    // workspace layout
    char* ws = (char*)d_ws;
    _Float16* Wh   = (_Float16*)(ws);                        // 0.94 MB
    _Float16* cbh  = (_Float16*)(ws + (size_t)1 * 1048576);  // 0.47 MB
    int*      amax = (int*)     (ws + (size_t)3 * 524288);   // 0.25 MB
    _Float16* prob = (_Float16*)(ws + (size_t)2 * 1048576);  // 40 MB

    cvt_f16_kernel<<<(NW_ + 255) / 256, 256, 0, stream>>>(Wf, cbf, Wh, cbh);

    dim3 g1(T_ / 128, GV_ / 128, B_);                // (32, 5, 8)
    gemm_logits_kernel<<<g1, 256, 0, stream>>>(inp, Wh, bias, logits);

    dim3 g2(T_ / 256, B_ * G_);                      // (16, 16)
    softmax_kernel<<<g2, 256, 0, stream>>>(logits, prob, amax);

    dim3 g3(T_ / 128, CG_ / 128, B_ * G_);           // (32, 3, 16)
    gemm_codes_kernel<<<g3, 256, 0, stream>>>(cbh, prob, cbf, amax, out);
}
